// LSTMModel_29953101923163
// MI455X (gfx1250) — matrix operand order repaired and checked by reading, not compiled, not run on hardware
//
#include <hip/hip_runtime.h>
#include <hip/hip_bf16.h>
#include <math.h>

#define B_SZ 256
#define T_SZ 1024
#define I_SZ 64
#define H_SZ 256
#define G_SZ 1024   // 4*H

typedef __attribute__((ext_vector_type(16))) __bf16 v16bf;
typedef __attribute__((ext_vector_type(8)))  float  v8f;

union FragU { uint4 u4[2]; v16bf v; };

// A fragment (16x32 bf16) from LDS, row-major [16 x stride].
// ISA layout: lane L holds row m=L%16, K = {k0..k0+7, k0+16..k0+23}, k0 = (L<16?0:8)+32*kb
// -> two contiguous 16B chunks per lane (ds_load_b128 x2).
__device__ __forceinline__ v16bf load_a_lds(const __bf16* s, int stride, int kb, int lane) {
    int m  = lane & 15;
    int k0 = kb * 32 + ((lane & 16) ? 8 : 0);
    const __bf16* p = s + m * stride + k0;
    FragU f;
    f.u4[0] = *(const uint4*)(p);
    f.u4[1] = *(const uint4*)(p + 16);
    return f.v;
}

// B fragment (32x16 bf16) from global bf16 weights stored [N rows][K] row-major
// (i.e. W itself, since B = W^T: B[k][n] = W[n][k]).
// ISA layout: lane L holds col n=L%16, K = {k0..k0+7, k0+16..k0+23}, k0 = (L<16?0:8)+32*kb
// -> two 16B chunks per lane (global_load_b128 x2).
__device__ __forceinline__ v16bf load_b_glb(const __bf16* w, int Kdim, int nbase, int kb, int lane) {
    int n  = nbase + (lane & 15);
    int k0 = kb * 32 + ((lane & 16) ? 8 : 0);
    const __bf16* p = w + (size_t)n * Kdim + k0;
    FragU f;
    f.u4[0] = *(const uint4*)(p);
    f.u4[1] = *(const uint4*)(p + 16);
    return f.v;
}

__global__ void cvt_weights_bf16(const float* __restrict__ wih_f,
                                 const float* __restrict__ whh_f,
                                 __bf16* __restrict__ wih_b,
                                 __bf16* __restrict__ whh_b) {
    int idx = blockIdx.x * blockDim.x + threadIdx.x;
    if (idx < G_SZ * I_SZ) wih_b[idx] = (__bf16)wih_f[idx];
    if (idx < G_SZ * H_SZ) whh_b[idx] = (__bf16)whh_f[idx];
}

// One workgroup per 16 batch rows; runs the full T=1024 recurrence.
// 1024 threads = 32 waves; wave w owns gate columns [32w, 32w+32).
__global__ void lstm_fused_kernel(const float* __restrict__ x,      // [B,T,I] fp32
                                  const __bf16* __restrict__ wih,   // [4H][I] bf16
                                  const __bf16* __restrict__ whh,   // [4H][H] bf16
                                  const float* __restrict__ b_ih,   // [4H]
                                  const float* __restrict__ b_hh,   // [4H]
                                  const float* __restrict__ fc_w,   // [H] (O=1)
                                  const float* __restrict__ fc_b,   // [1]
                                  float* __restrict__ out)          // [B]
{
    __shared__ __align__(16) __bf16 sh_h[16 * H_SZ];    // bf16 h operand   (8 KB)
    __shared__ __align__(16) __bf16 sh_x[16 * I_SZ];    // bf16 x tile      (2 KB)
    __shared__ __align__(16) float  sh_gates[16 * G_SZ];// fp32 gate preact (64 KB)
    __shared__ float sh_c[16 * H_SZ];                   // fp32 cell state  (16 KB)
    __shared__ float sh_hf[16 * H_SZ];                  // fp32 master h    (16 KB)
    __shared__ float sh_bias[G_SZ];                     // b_ih + b_hh      (4 KB)

    const int tid  = threadIdx.x;
    const int lane = tid & 31;
    const int wave = tid >> 5;           // 0..31
    const int b0   = blockIdx.x * 16;    // first batch row of this block
    const int ncol = wave * 32;          // first gate column owned by this wave

    // ---- init state ----
    sh_bias[tid] = b_ih[tid] + b_hh[tid];
    for (int u = tid; u < 16 * H_SZ; u += 1024) {
        sh_c[u]  = 0.f;
        sh_hf[u] = 0.f;
        sh_h[u]  = (__bf16)0.f;
    }
    __syncthreads();

    const int xm = tid >> 6;             // 0..15 batch row within block
    const int xi = tid & 63;             // 0..63 input feature
    const size_t xrow = (size_t)(b0 + xm) * T_SZ;

    const int cn = lane & 15;            // C/D: N = cn within tile
    const int mb = (lane & 16) ? 8 : 0;  // C/D: M base per lane half

    for (int t = 0; t < T_SZ; ++t) {
        // Opaque zero, re-materialized every iteration: makes the weight base
        // pointers loop-VARIANT in the IR so LICM cannot hoist the B-fragment
        // loads out of the t-loop (which previously caused ~160 VGPRs of
        // live weight fragments to spill to scratch). The loads stay ordinary
        // global_load_b128s served from L2 every step; cost = 1 s_mov/step.
        int opq;
        asm volatile("s_mov_b32 %0, 0" : "=s"(opq));
        const __bf16* wih_t = wih + opq;
        const __bf16* whh_t = whh + opq;

        // stage x[:,t,:] tile as bf16 (coalesced 64-float rows)
        float xv = x[(xrow + t) * I_SZ + xi];
        sh_x[xm * I_SZ + xi] = (__bf16)xv;
        if (t + 1 < T_SZ)   // prefetch next timestep's row (global_prefetch_b8)
            __builtin_prefetch(&x[(xrow + t + 1) * I_SZ + xi], 0, 0);
        __syncthreads();

        v8f acc0 = {};
        v8f acc1 = {};

        // input contribution: [16,64] x [64,32-cols]  (2 k-blocks)
#pragma unroll
        for (int kb = 0; kb < 2; ++kb) {
            v16bf a  = load_a_lds(sh_x, I_SZ, kb, lane);
            v16bf w0 = load_b_glb(wih_t, I_SZ, ncol,      kb, lane);
            v16bf w1 = load_b_glb(wih_t, I_SZ, ncol + 16, kb, lane);
            acc0 = __builtin_amdgcn_wmma_f32_16x16x32_bf16(false, a, false, w0,
                                                           (short)0, acc0, false, false);
            acc1 = __builtin_amdgcn_wmma_f32_16x16x32_bf16(false, a, false, w1,
                                                           (short)0, acc1, false, false);
        }
        // recurrent contribution: [16,256] x [256,32-cols] (8 k-blocks)
#pragma unroll
        for (int kb = 0; kb < 8; ++kb) {
            v16bf a  = load_a_lds(sh_h, H_SZ, kb, lane);
            v16bf w0 = load_b_glb(whh_t, H_SZ, ncol,      kb, lane);
            v16bf w1 = load_b_glb(whh_t, H_SZ, ncol + 16, kb, lane);
            acc0 = __builtin_amdgcn_wmma_f32_16x16x32_bf16(false, a, false, w0,
                                                           (short)0, acc0, false, false);
            acc1 = __builtin_amdgcn_wmma_f32_16x16x32_bf16(false, a, false, w1,
                                                           (short)0, acc1, false, false);
        }

        // scatter C tiles to LDS gate buffer (fp32)
#pragma unroll
        for (int r = 0; r < 8; ++r) {
            sh_gates[(mb + r) * G_SZ + ncol + cn]      = acc0[r];
            sh_gates[(mb + r) * G_SZ + ncol + 16 + cn] = acc1[r];
        }
        __syncthreads();

        // fused cell update, fp32 (4 hidden units per thread)
        for (int u = tid; u < 16 * H_SZ; u += 1024) {
            int m = u >> 8;
            int n = u & 255;
            const float* gr = &sh_gates[m * G_SZ];
            float gi = gr[n]       + sh_bias[n];
            float gf = gr[n + 256] + sh_bias[n + 256];
            float gg = gr[n + 512] + sh_bias[n + 512];
            float go = gr[n + 768] + sh_bias[n + 768];
            float iv = 1.f / (1.f + __expf(-gi));
            float fv = 1.f / (1.f + __expf(-gf));
            float gv = tanhf(gg);
            float ov = 1.f / (1.f + __expf(-go));
            float c  = fv * sh_c[u] + iv * gv;
            sh_c[u]  = c;
            float h  = ov * tanhf(c);
            sh_hf[u] = h;
            sh_h[u]  = (__bf16)h;   // layout m*H + n identical
        }
        __syncthreads();
    }

    // final FC (O=1): wave w reduces batch row w
    if (wave < 16) {
        float p = 0.f;
#pragma unroll
        for (int n = lane; n < H_SZ; n += 32)
            p += sh_hf[wave * H_SZ + n] * fc_w[n];
        p += __shfl_down(p, 16, 32);
        p += __shfl_down(p,  8, 32);
        p += __shfl_down(p,  4, 32);
        p += __shfl_down(p,  2, 32);
        p += __shfl_down(p,  1, 32);
        if (lane == 0) out[b0 + wave] = p + fc_b[0];
    }
}

extern "C" void kernel_launch(void* const* d_in, const int* in_sizes, int n_in,
                              void* d_out, int out_size, void* d_ws, size_t ws_size,
                              hipStream_t stream) {
    const float* x    = (const float*)d_in[0];
    const float* W_ih = (const float*)d_in[1];
    const float* W_hh = (const float*)d_in[2];
    const float* b_ih = (const float*)d_in[3];
    const float* b_hh = (const float*)d_in[4];
    const float* fc_w = (const float*)d_in[5];
    const float* fc_b = (const float*)d_in[6];
    float* out = (float*)d_out;

    __bf16* wih_b = (__bf16*)d_ws;                                          // 128 KB
    __bf16* whh_b = (__bf16*)((char*)d_ws + (size_t)G_SZ * I_SZ * 2);       // 512 KB

    cvt_weights_bf16<<<(G_SZ * H_SZ + 255) / 256, 256, 0, stream>>>(W_ih, W_hh, wih_b, whh_b);
    lstm_fused_kernel<<<B_SZ / 16, 1024, 0, stream>>>(x, wih_b, whh_b,
                                                      b_ih, b_hh, fc_w, fc_b, out);
}
